// GraphSAGE_13606456394538
// MI455X (gfx1250) — matrix operand order repaired and checked by reading, not compiled, not run on hardware
//
#include <hip/hip_runtime.h>
#include <stdint.h>

typedef __attribute__((ext_vector_type(16))) __bf16 v16bf;
typedef __attribute__((ext_vector_type(8)))  float  v8f;
typedef __attribute__((ext_vector_type(4)))  int    v4i;

#define N_NODES 50000
#define N_EDGES 400000
#define D_FEAT  512      // input dim of every scatter (512 for all 3 layers)
#define D_OUT3  128
#define KTOT    1024     // fused K: [mean | self] each 512

// Async global->LDS path (CDNA5 GLOBAL_LOAD_ASYNC_TO_LDS_B128, ASYNCcnt)
#if __has_builtin(__builtin_amdgcn_global_load_async_to_lds_b128)
#define USE_ASYNC_TILES 1
#define ASYNC_B128(g, l, OFF)                                                  \
  __builtin_amdgcn_global_load_async_to_lds_b128(                              \
      (__attribute__((address_space(1))) v4i*)(uintptr_t)(g),                  \
      (__attribute__((address_space(3))) v4i*)(uint32_t)(uintptr_t)(l),        \
      OFF, 0)
#else
#define USE_ASYNC_TILES 0
#endif

__device__ __forceinline__ void wait_async_all() {
#if __has_builtin(__builtin_amdgcn_s_wait_asynccnt)
  __builtin_amdgcn_s_wait_asynccnt(0);
#else
  asm volatile("s_wait_asynccnt 0x0" ::: "memory");
#endif
}

__device__ __forceinline__ unsigned short f32_to_bf16_bits(float f) {
  unsigned int u = __float_as_uint(f);
  unsigned int r = (u + 0x7FFFu + ((u >> 16) & 1u)) >> 16;   // RNE
  return (unsigned short)r;
}
__device__ __forceinline__ float bf16_bits_to_f32(unsigned short h) {
  return __uint_as_float(((unsigned int)h) << 16);
}

// ---------------- elementwise converts ----------------
__global__ void k_f32_to_bf16(const float* __restrict__ src,
                              unsigned short* __restrict__ dst, int n) {
  int i = blockIdx.x * blockDim.x + threadIdx.x;
  if (i < n) dst[i] = f32_to_bf16_bits(src[i]);
}

// W [K][Nd] f32 row-major  ->  Wt [Nd][KTOT] bf16, slice at kOff
__global__ void k_weightT_bf16(const float* __restrict__ W,
                               unsigned short* __restrict__ Wt,
                               int K, int Nd, int kOff) {
  int i = blockIdx.x * blockDim.x + threadIdx.x;
  if (i >= K * Nd) return;
  int k = i / Nd, n = i % Nd;
  Wt[(size_t)n * KTOT + kOff + k] = f32_to_bf16_bits(W[(size_t)k * Nd + n]);
}

// ---------------- graph aggregation ----------------
__global__ void k_degree(const int* __restrict__ dst, float* __restrict__ deg, int E) {
  int e = blockIdx.x * blockDim.x + threadIdx.x;
  if (e < E) atomicAdd(&deg[dst[e]], 1.0f);
}

template <bool SRC_BF16>
__global__ void k_scatter(const int* __restrict__ src, const int* __restrict__ dst,
                          const void* __restrict__ feat, float* __restrict__ agg, int E) {
  int idx = blockIdx.x * blockDim.x + threadIdx.x;
  if (idx >= E * (D_FEAT / 4)) return;
  int e = idx >> 7;            // D_FEAT/4 = 128 chunks per edge
  int f = (idx & 127) << 2;    // 4 features per thread
  int s = src[e], d = dst[e];
  float v0, v1, v2, v3;
  if (SRC_BF16) {
    const unsigned short* p = (const unsigned short*)feat + (size_t)s * D_FEAT + f;
    uint2 raw = *(const uint2*)p;
    v0 = bf16_bits_to_f32((unsigned short)(raw.x & 0xFFFFu));
    v1 = bf16_bits_to_f32((unsigned short)(raw.x >> 16));
    v2 = bf16_bits_to_f32((unsigned short)(raw.y & 0xFFFFu));
    v3 = bf16_bits_to_f32((unsigned short)(raw.y >> 16));
  } else {
    float4 raw = *(const float4*)((const float*)feat + (size_t)s * D_FEAT + f);
    v0 = raw.x; v1 = raw.y; v2 = raw.z; v3 = raw.w;
  }
  float* a = agg + (size_t)d * D_FEAT + f;   // 102 MB accumulator lives in L2
  atomicAdd(a + 0, v0); atomicAdd(a + 1, v1);
  atomicAdd(a + 2, v2); atomicAdd(a + 3, v3);
}

__global__ void k_mean_bf16(const float* __restrict__ agg, const float* __restrict__ deg,
                            unsigned short* __restrict__ mean, int n) {
  int i = blockIdx.x * blockDim.x + threadIdx.x;
  if (i >= n * D_FEAT) return;
  int row = i >> 9;            // D_FEAT = 512
  float d = fmaxf(deg[row], 1.0f);
  mean[i] = f32_to_bf16_bits(agg[i] / d);
}

// ---------------- fused SAGE GEMM: C = [mean|self] @ Wt^T + b ----------------
// 256 threads (8 waves), 128x128 C tile, K stepped by 32 through double-buffered
// LDS staged with async global->LDS copies. Wave (wm,wn) owns 32x64 = 2x4 WMMAs.
template <bool RELU_BF16_OUT>
__global__ __launch_bounds__(256)
void k_sage_gemm(const unsigned short* __restrict__ Amean,   // [N][512] bf16
                 const unsigned short* __restrict__ Aself,   // [N][512] bf16
                 const unsigned short* __restrict__ Wt,      // [dOut][KTOT] bf16
                 const float* __restrict__ bias,             // [dOut]
                 void* __restrict__ outv,
                 int nRows, int dOut) {
  __shared__ unsigned short As[2][128 * 40];   // 128 rows x 32 k, pad->40
  __shared__ unsigned short Bs[2][128 * 40];   // 128 cols x 32 k, pad->40

  const int tid  = threadIdx.x;
  const int lane = tid & 31;
  const int wave = tid >> 5;
  const int wm = wave & 3;       // 4 waves over M (32 rows each)
  const int wn = wave >> 2;      // 2 waves over N (64 cols each)
  const int lm = lane & 15;
  const int hh = lane >> 4;      // lane half selects K sub-block
  const int mBase = blockIdx.x * 128;
  const int nBase = blockIdx.y * 128;

  v8f acc[2][4];
  for (int i = 0; i < 2; ++i)
    for (int j = 0; j < 4; ++j)
      for (int e = 0; e < 8; ++e) acc[i][j][e] = 0.0f;

  const int lr = tid >> 1;          // 0..127 : tile row / tile col
  const int lc = (tid & 1) * 16;    // 0 or 16 : k sub-chunk

  int growA = mBase + lr; if (growA >= nRows) growA = nRows - 1;

  // stage a 128x32 A tile + 128x32 B tile into LDS buffer `stage`
  auto tile_load = [&](int stage, int kb) {
    const unsigned short* Asrc = (kb < 512) ? Amean : Aself;
    const int kloc = kb & 511;
    const unsigned short* ga = Asrc + (size_t)growA * 512 + kloc + lc;
    const unsigned short* gb = Wt + (size_t)(nBase + lr) * KTOT + kb + lc;
    unsigned short* la = &As[stage][lr * 40 + lc];
    unsigned short* lb = &Bs[stage][lr * 40 + lc];
#if USE_ASYNC_TILES
    ASYNC_B128(ga, la, 0);
    ASYNC_B128(ga, la, 16);
    ASYNC_B128(gb, lb, 0);
    ASYNC_B128(gb, lb, 16);
#else
    uint4 a0 = ((const uint4*)ga)[0];
    uint4 a1 = ((const uint4*)ga)[1];
    uint4 b0 = ((const uint4*)gb)[0];
    uint4 b1 = ((const uint4*)gb)[1];
    *(uint4*)(la)     = a0;
    *(uint4*)(la + 8) = a1;
    *(uint4*)(lb)     = b0;
    *(uint4*)(lb + 8) = b1;
#endif
  };

  tile_load(0, 0);

  for (int it = 0; it < KTOT / 32; ++it) {
    const int kb  = it * 32;
    const int cur = it & 1;
#if USE_ASYNC_TILES
    wait_async_all();               // DMA for stage `cur` complete (this wave)
#endif
    __syncthreads();                // all waves' tile `cur` visible
    if (kb + 32 < KTOT) tile_load(cur ^ 1, kb + 32);   // overlap next stage

    union Frag { v16bf v; unsigned int u[8]; } af[2], bfr[4];
    // A frags: ISA 16-bit A 16x32 layout (lanes 0-15 K:0-7/16-23, lanes 16-31 K:8-15/24-31)
    for (int i = 0; i < 2; ++i) {
      const unsigned short* base = &As[cur][(wm * 32 + i * 16 + lm) * 40];
#pragma unroll
      for (int j = 0; j < 4; ++j)
        af[i].u[j] = *(const unsigned int*)&base[8 * hh + 2 * j];
#pragma unroll
      for (int j = 4; j < 8; ++j)
        af[i].u[j] = *(const unsigned int*)&base[16 + 8 * hh + 2 * (j - 4)];
    }
    // B frags: same K order as A (lanes 0-15 K:0-7/16-23, lanes 16-31 K:8-15/24-31), N = lane&15
    for (int j2 = 0; j2 < 4; ++j2) {
      const unsigned short* base = &Bs[cur][(wn * 64 + j2 * 16 + lm) * 40];
#pragma unroll
      for (int j = 0; j < 8; ++j)
        bfr[j2].u[j] = *(const unsigned int*)&base[(j < 4) ? (8 * hh + 2 * j) : (16 + 8 * hh + 2 * (j - 4))];
    }
#pragma unroll
    for (int i = 0; i < 2; ++i)
#pragma unroll
      for (int j2 = 0; j2 < 4; ++j2)
        acc[i][j2] = __builtin_amdgcn_wmma_f32_16x16x32_bf16(
            false, af[i].v, false, bfr[j2].v, (short)0, acc[i][j2], false, false);
  }

  // epilogue: bias (+ReLU, bf16) or f32 store; C layout: VGPR r -> M = r + 8*hh, N = lane&15
  for (int i = 0; i < 2; ++i) {
    for (int j2 = 0; j2 < 4; ++j2) {
      const int gcol = nBase + wn * 64 + j2 * 16 + lm;
      const float bv = bias[gcol];
#pragma unroll
      for (int r = 0; r < 8; ++r) {
        const int grow = mBase + wm * 32 + i * 16 + hh * 8 + r;
        if (grow < nRows) {
          float v = acc[i][j2][r] + bv;
          if (RELU_BF16_OUT) {
            v = fmaxf(v, 0.0f);
            ((unsigned short*)outv)[(size_t)grow * dOut + gcol] = f32_to_bf16_bits(v);
          } else {
            ((float*)outv)[(size_t)grow * dOut + gcol] = v;
          }
        }
      }
    }
  }
}

// ---------------- log-softmax over 128 classes, one block per row ----------------
__global__ void k_log_softmax(float* __restrict__ out) {
  __shared__ float red[128];
  const int row = blockIdx.x;
  const int t = threadIdx.x;
  float v = out[(size_t)row * D_OUT3 + t];
  red[t] = v; __syncthreads();
  for (int s = 64; s > 0; s >>= 1) {
    if (t < s) red[t] = fmaxf(red[t], red[t + s]);
    __syncthreads();
  }
  const float m = red[0]; __syncthreads();
  red[t] = __expf(v - m); __syncthreads();
  for (int s = 64; s > 0; s >>= 1) {
    if (t < s) red[t] += red[t + s];
    __syncthreads();
  }
  const float lse = __logf(red[0]);
  out[(size_t)row * D_OUT3 + t] = v - m - lse;
}

// ---------------- launcher ----------------
extern "C" void kernel_launch(void* const* d_in, const int* in_sizes, int n_in,
                              void* d_out, int out_size, void* d_ws, size_t ws_size,
                              hipStream_t stream) {
  (void)in_sizes; (void)n_in; (void)out_size; (void)ws_size;
  const float* x   = (const float*)d_in[0];
  const int*   ei  = (const int*)d_in[1];
  const float* W1l = (const float*)d_in[2];
  const float* b1  = (const float*)d_in[3];
  const float* W1r = (const float*)d_in[4];
  const float* W2l = (const float*)d_in[5];
  const float* b2  = (const float*)d_in[6];
  const float* W2r = (const float*)d_in[7];
  const float* W3l = (const float*)d_in[8];
  const float* b3  = (const float*)d_in[9];
  const float* W3r = (const float*)d_in[10];
  const int* srcI = ei;
  const int* dstI = ei + N_EDGES;

  // workspace carve-up (~308 MB)
  auto al = [](size_t v) { return (v + 255) & ~(size_t)255; };
  char* w = (char*)d_ws;
  float* deg            = (float*)w;          w += al((size_t)N_NODES * 4);
  float* agg            = (float*)w;          w += al((size_t)N_NODES * D_FEAT * 4);
  unsigned short* meanB = (unsigned short*)w; w += al((size_t)N_NODES * D_FEAT * 2);
  unsigned short* xB    = (unsigned short*)w; w += al((size_t)N_NODES * D_FEAT * 2);
  unsigned short* h1B   = (unsigned short*)w; w += al((size_t)N_NODES * D_FEAT * 2);
  unsigned short* h2B   = (unsigned short*)w; w += al((size_t)N_NODES * D_FEAT * 2);
  unsigned short* wtB   = (unsigned short*)w; w += al((size_t)512 * KTOT * 2);

  const int T = 256;
  const int nElem  = N_NODES * D_FEAT;                 // 25.6M
  const int nScat  = N_EDGES * (D_FEAT / 4);           // 51.2M
  const dim3 gGemm((N_NODES + 127) / 128, 4);
  const dim3 gGemm3((N_NODES + 127) / 128, 1);

  // degree (shared by all layers) + x -> bf16
  (void)hipMemsetAsync(deg, 0, (size_t)N_NODES * 4, stream);
  k_degree<<<(N_EDGES + T - 1) / T, T, 0, stream>>>(dstI, deg, N_EDGES);
  k_f32_to_bf16<<<(nElem + T - 1) / T, T, 0, stream>>>(x, xB, nElem);

  // ---- layer 1 ----
  (void)hipMemsetAsync(agg, 0, (size_t)N_NODES * D_FEAT * 4, stream);
  k_scatter<false><<<(nScat + T - 1) / T, T, 0, stream>>>(srcI, dstI, x, agg, N_EDGES);
  k_mean_bf16<<<(nElem + T - 1) / T, T, 0, stream>>>(agg, deg, meanB, N_NODES);
  k_weightT_bf16<<<(512 * 512 + T - 1) / T, T, 0, stream>>>(W1l, wtB, 512, 512, 0);
  k_weightT_bf16<<<(512 * 512 + T - 1) / T, T, 0, stream>>>(W1r, wtB, 512, 512, 512);
  k_sage_gemm<true><<<gGemm, T, 0, stream>>>(meanB, xB, wtB, b1, h1B, N_NODES, 512);

  // ---- layer 2 ----
  (void)hipMemsetAsync(agg, 0, (size_t)N_NODES * D_FEAT * 4, stream);
  k_scatter<true><<<(nScat + T - 1) / T, T, 0, stream>>>(srcI, dstI, h1B, agg, N_EDGES);
  k_mean_bf16<<<(nElem + T - 1) / T, T, 0, stream>>>(agg, deg, meanB, N_NODES);
  k_weightT_bf16<<<(512 * 512 + T - 1) / T, T, 0, stream>>>(W2l, wtB, 512, 512, 0);
  k_weightT_bf16<<<(512 * 512 + T - 1) / T, T, 0, stream>>>(W2r, wtB, 512, 512, 512);
  k_sage_gemm<true><<<gGemm, T, 0, stream>>>(meanB, h1B, wtB, b2, h2B, N_NODES, 512);

  // ---- layer 3 + log_softmax ----
  (void)hipMemsetAsync(agg, 0, (size_t)N_NODES * D_FEAT * 4, stream);
  k_scatter<true><<<(nScat + T - 1) / T, T, 0, stream>>>(srcI, dstI, h2B, agg, N_EDGES);
  k_mean_bf16<<<(nElem + T - 1) / T, T, 0, stream>>>(agg, deg, meanB, N_NODES);
  k_weightT_bf16<<<(512 * 128 + T - 1) / T, T, 0, stream>>>(W3l, wtB, 512, 128, 0);
  k_weightT_bf16<<<(512 * 128 + T - 1) / T, T, 0, stream>>>(W3r, wtB, 512, 128, 512);
  k_sage_gemm<false><<<gGemm3, T, 0, stream>>>(meanB, h2B, wtB, b3, (float*)d_out,
                                               N_NODES, 128);
  k_log_softmax<<<N_NODES, 128, 0, stream>>>((float*)d_out);
}
